// CustomRNNCell_4982162063621
// MI455X (gfx1250) — compile-verified, matrix operand order checked by reading, not run on hardware
//
#include <hip/hip_runtime.h>
#include <math.h>

typedef __attribute__((ext_vector_type(16))) __bf16 v16bf;
typedef __attribute__((ext_vector_type(8)))  __bf16 v8bf;
typedef __attribute__((ext_vector_type(8)))  float  v8f;

#define T_   512
#define U_   128
#define N_   80
#define H_   512
#define M_   10
#define KTOT 608   // 96 (x|w|pad) + 512 (h)
#define KT_  19    // KTOT / 32
#define NT_  128   // 2048 / 16
#define AST  608   // A-panel row stride (bf16 elems); 1216B rows, 16B aligned

// ---------------------------------------------------------------------------
// Prep: pack [Wx ; 0-pad ; Wh] (608 x 2048, fp32) into bf16 WMMA B-fragments.
// Fragment f = nt*KT_ + kt holds the 32x16 tile (K = kt*32.., N = nt*16..) in
// the exact per-lane register order of V_WMMA_*_16X16X32_BF16's B operand:
// lane l -> column n = nt*16 + (l&15), K-halves selected by l>>4; 16 bf16
// contiguous per lane => the main kernel loads one v16bf per lane.
// ---------------------------------------------------------------------------
__global__ __launch_bounds__(256) void pack_weights(const float* __restrict__ Wx,
                                                    const float* __restrict__ Wh,
                                                    __bf16* __restrict__ wpack) {
    int f  = blockIdx.x;           // 0 .. NT_*KT_-1
    int nt = f / KT_, kt = f % KT_;
    for (int e = threadIdx.x; e < 512; e += 256) {
        int lane = e >> 4, j = e & 15;
        int n  = nt * 16 + (lane & 15);
        int hf = lane >> 4;
        int kk = (j < 8) ? (hf * 8 + j) : (16 + hf * 8 + (j - 8));
        int k  = kt * 32 + kk;
        float v = 0.f;
        if (k < 83)       v = Wx[k * 2048 + n];          // inp rows: x(3) + w(80)
        else if (k >= 96) v = Wh[(k - 96) * 2048 + n];   // recurrent rows
        wpack[(size_t)f * 512 + e] = (__bf16)v;
    }
}

__device__ __forceinline__ float sigf(float x) { return 1.f / (1.f + __expf(-x)); }

// ---------------------------------------------------------------------------
// Persistent kernel: one workgroup per 16-batch tile, full T=512 scan inside.
// 8 waves; wave w owns h-dims [64w, 64w+64) => 4 column quadruples (i,f,g,o).
// LSTM cell state c lives in WMMA-accumulator-layout VGPRs for the whole scan.
// ---------------------------------------------------------------------------
__global__ __launch_bounds__(256) void rnn_attn(const float* __restrict__ strokes,
                                                const float* __restrict__ trans,
                                                const float* __restrict__ bias,
                                                const float* __restrict__ Wd,
                                                const float* __restrict__ bd,
                                                const __bf16* __restrict__ wpack,
                                                float* __restrict__ out) {
    __shared__ __align__(16) __bf16 sA[2][16][AST];  // double-buffered A panel
    __shared__ float sY[16][30];                     // exp(h@Wd + bd)
    __shared__ float sWf[16][132];                   // phi summed over mixtures
    __shared__ float sKap[16][M_];                   // running kappa

    const int tid  = threadIdx.x;
    const int wv   = tid >> 5;
    const int lane = tid & 31;
    const int mrow = lane & 15;
    const int hf   = lane >> 4;
    const int bg   = blockIdx.x * 16;                // batch base

    // ---- init: zero both A buffers (pad cols 83..95 stay 0 forever), kappa=0
    {
        __bf16* p = &sA[0][0][0];
        for (int i = tid; i < 2 * 16 * AST; i += 256) p[i] = (__bf16)0.f;
        if (tid < 16 * M_) sKap[tid / M_][tid % M_] = 0.f;
    }
    __syncthreads();
    if (tid < 48) {   // x_0 into buffer 0
        int b = tid / 3, c = tid % 3;
        sA[0][b][c] = (__bf16)strokes[((size_t)(bg + b) * T_) * 3 + c];
    }
    __syncthreads();

    v8f cst[4] = {};  // persistent LSTM cell state (accumulator layout)

    for (int t = 0; t < T_; ++t) {
        const int cur = t & 1, nxt = cur ^ 1;
        const __bf16* aRow = &sA[cur][mrow][hf * 8];
        const v16bf* wp = (const v16bf*)wpack;

        #pragma unroll 1
        for (int q = 0; q < 4; ++q) {
            const int nt0 = wv * 4 + q;              // i-gate column tile
            v8f ai = {}, af = {}, ag = {}, ao = {};
            for (int kt = 0; kt < KT_; ++kt) {
                v8bf lo = *(const v8bf*)(aRow + kt * 32);
                v8bf hi = *(const v8bf*)(aRow + kt * 32 + 16);
                v16bf a = __builtin_shufflevector(lo, hi,
                            0,1,2,3,4,5,6,7,8,9,10,11,12,13,14,15);
                v16bf fi = wp[((nt0     ) * KT_ + kt) * 32 + lane];
                v16bf ff = wp[((nt0 + 32) * KT_ + kt) * 32 + lane];
                v16bf fg = wp[((nt0 + 64) * KT_ + kt) * 32 + lane];
                v16bf fo = wp[((nt0 + 96) * KT_ + kt) * 32 + lane];
                ai = __builtin_amdgcn_wmma_f32_16x16x32_bf16(false, a, false, fi, (short)0, ai, false, false);
                af = __builtin_amdgcn_wmma_f32_16x16x32_bf16(false, a, false, ff, (short)0, af, false, false);
                ag = __builtin_amdgcn_wmma_f32_16x16x32_bf16(false, a, false, fg, (short)0, ag, false, false);
                ao = __builtin_amdgcn_wmma_f32_16x16x32_bf16(false, a, false, fo, (short)0, ao, false, false);
            }
            const int col = wv * 64 + q * 16 + (lane & 15);
            const float Bi = bias[col], Bf = bias[512 + col];
            const float Bg = bias[1024 + col], Bo = bias[1536 + col];
            v8f cv = cst[q];
            #pragma unroll
            for (int r = 0; r < 8; ++r) {
                int mm = hf * 8 + r;                  // batch row in tile
                float iv = sigf(ai[r] + Bi);
                float fv = sigf(af[r] + Bf);
                float gv = tanhf(ag[r] + Bg);
                float ov = sigf(ao[r] + Bo);
                float cn = fv * cv[r] + iv * gv;
                cv[r] = cn;
                float hn = ov * tanhf(cn);
                sA[nxt][mm][96 + col] = (__bf16)hn;   // h for next step's GEMM
                out[((size_t)(bg + mm) * T_ + t) * 593 + col] = hn;
            }
            cst[q] = cv;
        }
        __syncthreads();

        // ---- attention head: y = exp(h @ Wd + bd)   (16x512 @ 512x30)
        for (int item = tid; item < 16 * 30; item += 256) {
            int b = item / 30, cc = item % 30;
            float dot = 0.f;
            const __bf16* hrow = &sA[nxt][b][96];
            for (int k = 0; k < H_; ++k) dot += (float)hrow[k] * Wd[k * 30 + cc];
            sY[b][cc] = __expf(dot + bd[cc]);
        }
        __syncthreads();
        if (tid < 16 * M_) sKap[tid / M_][tid % M_] += sY[tid / M_][20 + tid % M_];
        __syncthreads();
        // ---- phi over U+1 positions, summed over mixtures
        for (int item = tid; item < 16 * (U_ + 1); item += 256) {
            int b = item / (U_ + 1), u = item % (U_ + 1);
            float s = 0.f;
            #pragma unroll
            for (int mi = 0; mi < M_; ++mi) {
                float d = sKap[b][mi] - (float)u;
                s += sY[b][mi] * __expf(-sY[b][10 + mi] * d * d);
            }
            sWf[b][u] = s;
        }
        __syncthreads();
        // ---- argmax (first max, matching jnp.argmax)
        if (tid < 16) {
            int b = tid; float mx = sWf[b][0]; int mi = 0;
            for (int u = 1; u <= U_; ++u) {
                float v = sWf[b][u];
                if (v > mx) { mx = v; mi = u; }
            }
            out[((size_t)(bg + b) * T_ + t) * 593 + 592] = (float)mi;
        }
        // ---- w = wfull[:, :U] @ transcriptions; feed back into next A panel
        for (int item = tid; item < 16 * N_; item += 256) {
            int b = item / N_, n = item % N_;
            float s = 0.f;
            const float* tr = trans + ((size_t)(bg + b) * U_) * N_ + n;
            for (int u = 0; u < U_; ++u) s += sWf[b][u] * tr[u * N_];
            out[((size_t)(bg + b) * T_ + t) * 593 + 512 + n] = s;
            sA[nxt][b][3 + n] = (__bf16)s;
        }
        if (t + 1 < T_ && tid < 48) {                 // x_{t+1}
            int b = tid / 3, c = tid % 3;
            sA[nxt][b][c] = (__bf16)strokes[((size_t)(bg + b) * T_ + (t + 1)) * 3 + c];
        }
        __syncthreads();
    }
}

extern "C" void kernel_launch(void* const* d_in, const int* in_sizes, int n_in,
                              void* d_out, int out_size, void* d_ws, size_t ws_size,
                              hipStream_t stream) {
    (void)in_sizes; (void)n_in; (void)out_size; (void)ws_size;
    const float* strokes = (const float*)d_in[0];
    const float* trans   = (const float*)d_in[1];
    // d_in[2] = enumerated (arange 0..U) -- recomputed on the fly
    const float* Wx      = (const float*)d_in[3];
    const float* Wh      = (const float*)d_in[4];
    const float* bias    = (const float*)d_in[5];
    const float* Wd      = (const float*)d_in[6];
    const float* bd      = (const float*)d_in[7];
    float* out           = (float*)d_out;
    __bf16* wpack        = (__bf16*)d_ws;   // 128*19 fragments * 512 bf16 = ~2.4 MB

    pack_weights<<<NT_ * KT_, 256, 0, stream>>>(Wx, Wh, wpack);
    rnn_attn<<<8, 256, 0, stream>>>(strokes, trans, bias, Wd, bd, wpack, out);
}
